// DeformConv1d_39771397161672
// MI455X (gfx1250) — matrix operand order repaired and checked by reading, not compiled, not run on hardware
//
#include <hip/hip_runtime.h>

typedef __attribute__((ext_vector_type(16))) _Float16 v16h;
typedef __attribute__((ext_vector_type(8)))  _Float16 v8h;
typedef __attribute__((ext_vector_type(8)))  float    v8f;
typedef __attribute__((ext_vector_type(2)))  float    vf2;
typedef vf2 __attribute__((aligned(4)))      vf2u;    // 8B load, 4B alignment ok on gfx1250

#define L_IN   16384
#define L_OUT  16378
#define CIN    256
#define COUT   256
#define KTAP   7
#define RDIM   (CIN * KTAP)   // 1792 reduction length
#define KC     32             // reduction per WMMA
#define NKT    (RDIM / KC)    // 56 k-chunks
#define NB     64             // N columns per workgroup
#define NFRAG  (NKT * 16)     // 896 A fragments (16 m-tiles each chunk)
#define BUFH   (4 * 32 * 16)  // halves per LDS B buffer (2048)

// ---------------------------------------------------------------------------
// Pack W[o, c, k] (f32) into f16 A-fragments (16x32, 16-bit layout from ISA):
//   lanes 0-15: M=lane,   halves 0-7 -> K=0..7,  halves 8-15 -> K=16..23
//   lanes16-31: M=lane-16, K ranges shifted by +8
// Reduction index r = k*256 + c. Fragment id = kt*16 + mtile.
// ---------------------------------------------------------------------------
__global__ void pack_w_kernel(const float* __restrict__ w, _Float16* __restrict__ wsA) {
    int gt   = blockIdx.x * 256 + threadIdx.x;
    int frag = gt >> 5;
    int lane = gt & 31;
    if (frag >= NFRAG) return;
    int mtile = frag & 15;
    int kt    = frag >> 4;
    int o     = mtile * 16 + (lane & 15);
    _Float16* dst = wsA + ((size_t)frag * 32 + lane) * 16;
#pragma unroll
    for (int h = 0; h < 16; ++h) {
        int j = (h & 7) + ((h >> 3) << 4) + ((lane >> 4) << 3); // local K 0..31
        int r = kt * KC + j;
        int k = r >> 8;       // tap index
        int c = r & 255;      // input channel
        dst[h] = (_Float16)w[((size_t)o * CIN + c) * KTAP + k];
    }
}

// ---------------------------------------------------------------------------
// Fused interpolate + GEMM, double-buffered LDS (integer-offset buffer select
// so stores stay in the DS address space, not FLAT).
// One block = all 256 output rows x 64 columns; 8 waves, wave w owns
// m-tiles {2w, 2w+1} and all 4 n-tiles. One barrier per k-chunk.
// ---------------------------------------------------------------------------
__global__ __launch_bounds__(256)
void deform_gemm_kernel(const float* __restrict__ x,
                        const float* __restrict__ offs,
                        const _Float16* __restrict__ wsA,
                        const float* __restrict__ bias,
                        float* __restrict__ out) {
    __shared__ __align__(16) _Float16 ldsB[2 * BUFH]; // 2 x 4KB, flattened

    const int b  = blockIdx.y;
    const int l0 = blockIdx.x * NB;
    const int t  = threadIdx.x;

    // producer role: 64 columns x 4 groups of 8 channels
    const int n  = t & 63;           // column within slab
    const int jg = t >> 6;           // 0..3 -> local-K group of 8
    const int l  = l0 + n;
    const int lc = (l < L_OUT) ? l : (L_OUT - 1);
    const int ntile = n >> 4;
    const int ncol  = n & 15;

    // consumer role
    const int wave = t >> 5;
    const int lane = t & 31;

    // B-fragment LDS slot is constant per thread:
    //   lane = ncol + 16*((j>>3)&1), slot = (j&7) + 8*(j>>4); this thread owns j = jg*8..jg*8+7
    const int ldsoff = (((ntile * 32) + ncol + ((jg & 1) << 4)) << 4) + ((jg >> 1) << 3);

    const float* xb    = x + (size_t)b * CIN * L_IN;
    const float* offsb = offs + ((size_t)b * L_OUT + lc) * KTAP;

    v8f acc[2][4] = {};
    float w0 = 0.f, w1 = 0.f;
    int   U  = 0;

    // produce chunk p: one 8B gather + fma + cvt per channel, one ds_store_b128
    auto produce = [&](int p) {
        const int k = p >> 3;
        if ((p & 7) == 0) {
            float off = offsb[k];
            float T   = (float)lc + (float)k + off;
            T = fminf(fmaxf(T, (float)lc), (float)lc + (float)(KTAP - 1));
            float Uf = floorf(T);
            Uf = fminf(fmaxf(Uf, 0.f), (float)(L_IN - 2));
            U  = (int)Uf;
            w0 = fmaxf(0.f, 1.f - fabsf(Uf - T));
            w1 = fmaxf(0.f, 1.f - fabsf(Uf + 1.f - T));
        }
        const int c0 = ((p & 7) << 5) + (jg << 3);
        const float* base = xb + (size_t)c0 * L_IN + U;
        v8h hv;
#pragma unroll
        for (int i = 0; i < 8; ++i) {
            vf2u xv = *(const vf2u*)(base + (size_t)i * L_IN); // x[c,U], x[c,U+1]
            hv[i] = (_Float16)(w0 * xv.x + w1 * xv.y);
        }
        *(v8h*)&ldsB[((p & 1) ? BUFH : 0) + ldsoff] = hv;      // ds_store_b128
    };

    produce(0);
    __syncthreads();

    for (int kt = 0; kt < NKT; ++kt) {
        // A fragments for this chunk (contiguous 32B per lane, L2-resident)
        const v16h a0 = *(const v16h*)(wsA + (((size_t)(kt * 16 + wave * 2 + 0) * 32 + lane) << 4));
        const v16h a1 = *(const v16h*)(wsA + (((size_t)(kt * 16 + wave * 2 + 1) * 32 + lane) << 4));

        // overlap: produce next chunk into the other buffer while WMMAs run
        if (kt + 1 < NKT) produce(kt + 1);

        const int rbase = ((kt & 1) ? BUFH : 0) + (lane << 4);
#pragma unroll
        for (int nt = 0; nt < 4; ++nt) {
            const v16h bf = *(const v16h*)&ldsB[rbase + (nt << 9)];
            acc[0][nt] = __builtin_amdgcn_wmma_f32_16x16x32_f16(
                false, a0, false, bf, (short)0, acc[0][nt], false, false);
            acc[1][nt] = __builtin_amdgcn_wmma_f32_16x16x32_f16(
                false, a1, false, bf, (short)0, acc[1][nt], false, false);
        }
        __syncthreads();
    }

    // store: C/D layout -> VGPR i holds rows i (lanes 0-15) and i+8 (lanes 16-31).
    // Non-temporal: output is write-once, keep x/W hot in L2 instead.
    const int mrow_off = (lane >> 4) << 3;
    const int colsel   = lane & 15;
#pragma unroll
    for (int m = 0; m < 2; ++m) {
        const int mt = wave * 2 + m;
#pragma unroll
        for (int nt = 0; nt < 4; ++nt) {
            const int ll = l0 + nt * 16 + colsel;
            if (ll < L_OUT) {
#pragma unroll
                for (int i = 0; i < 8; ++i) {
                    const int o = mt * 16 + i + mrow_off;
                    __builtin_nontemporal_store(acc[m][nt][i] + bias[o],
                                                &out[((size_t)b * COUT + o) * L_OUT + ll]);
                }
            }
        }
    }
}

extern "C" void kernel_launch(void* const* d_in, const int* in_sizes, int n_in,
                              void* d_out, int out_size, void* d_ws, size_t ws_size,
                              hipStream_t stream) {
    const float* x    = (const float*)d_in[0];  // [2,256,16384]
    const float* offs = (const float*)d_in[1];  // [2,1,16378,7]
    const float* w    = (const float*)d_in[2];  // [256,256,7]
    const float* bias = (const float*)d_in[3];  // [256]
    float*       out  = (float*)d_out;          // [2,256,16378]
    _Float16*    wsA  = (_Float16*)d_ws;        // 896 frags * 1KB = ~0.9 MB

    {
        int threads = NFRAG * 32;
        pack_w_kernel<<<(threads + 255) / 256, 256, 0, stream>>>(w, wsA);
    }
    {
        dim3 grid((L_OUT + NB - 1) / NB, 2);    // 256 x 2 blocks
        deform_gemm_kernel<<<grid, 256, 0, stream>>>(x, offs, wsA, bias, out);
    }
}
